// PhysicsCrossAttention_90615220011105
// MI455X (gfx1250) — matrix operand order repaired and checked by reading, not compiled, not run on hardware
//
#include <hip/hip_runtime.h>
#include <hip/hip_bf16.h>
#include <math.h>

// ---------------------------------------------------------------------------
// Problem constants (from reference)
// ---------------------------------------------------------------------------
#define B_TOK   16384
#define D_MODEL 1024
#define PHYS_N  11
#define NT_TOK  4
#define NH_HEAD 4
#define DH_HEAD 256
#define DN_PROJ 4096   // D * NT

typedef __attribute__((ext_vector_type(16))) __bf16 v16bf;
typedef __attribute__((ext_vector_type(8)))  float  v8f;

// ---------------------------------------------------------------------------
// Scalar helpers (bf16 kept as raw bits to avoid any __bf16 scalar arithmetic)
// ---------------------------------------------------------------------------
__device__ __forceinline__ unsigned short f2bf_bits(float f) {
  unsigned u = __builtin_bit_cast(unsigned, f);
  unsigned r = u + 0x7FFFu + ((u >> 16) & 1u);   // round-to-nearest-even
  return (unsigned short)(r >> 16);
}
__device__ __forceinline__ float bf2f_bits(unsigned short s) {
  unsigned u = ((unsigned)s) << 16;
  return __builtin_bit_cast(float, u);
}
__device__ __forceinline__ float gelu_t(float x) {   // tanh approximation (jax default)
  float x3 = x * x * x;
  return 0.5f * x * (1.0f + tanhf(0.7978845608028654f * (x + 0.044715f * x3)));
}
__device__ __forceinline__ float loadval(const float* p) { return *p; }
__device__ __forceinline__ float loadval(const unsigned short* p) { return bf2f_bits(*p); }

// ---------------------------------------------------------------------------
// fp32 [K,N] -> bf16-bits [N,K] transposing conversion (weights, done once).
// ---------------------------------------------------------------------------
__global__ __launch_bounds__(256) void f2bf_transpose_kernel(const float* __restrict__ x,
                                                             unsigned short* __restrict__ y,
                                                             int K, int N) {
  __shared__ float tile[32][33];
  const int kb = blockIdx.y * 32;
  const int nb = blockIdx.x * 32;
  const int tx = threadIdx.x & 31;
  const int ty = threadIdx.x >> 5;   // 0..7
#pragma unroll
  for (int i = 0; i < 4; ++i)
    tile[ty + 8 * i][tx] = x[(size_t)(kb + ty + 8 * i) * N + nb + tx];
  __syncthreads();
#pragma unroll
  for (int i = 0; i < 4; ++i)
    y[(size_t)(nb + ty + 8 * i) * K + kb + tx] = f2bf_bits(tile[tx][ty + 8 * i]);
}

// ---------------------------------------------------------------------------
// phys stage 1: h = gelu(physics @ pw1 + pb1), K = 11 (scalar, memory bound)
// ---------------------------------------------------------------------------
__global__ __launch_bounds__(256) void phys_h_kernel(const float* __restrict__ physics,
                                                     const float* __restrict__ pw1,
                                                     const float* __restrict__ pb1,
                                                     unsigned short* __restrict__ h) {
  const int b = blockIdx.x;
  const int tid = threadIdx.x;
  __shared__ float ph[PHYS_N];
  if (tid < PHYS_N) ph[tid] = physics[(size_t)b * PHYS_N + tid];
  __syncthreads();
#pragma unroll
  for (int j = 0; j < DN_PROJ / 256; ++j) {
    int col = tid + j * 256;
    float acc = pb1[col];
#pragma unroll
    for (int p = 0; p < PHYS_N; ++p) acc += ph[p] * pw1[(size_t)p * DN_PROJ + col];
    h[(size_t)b * DN_PROJ + col] = f2bf_bits(gelu_t(acc));
  }
}

// ---------------------------------------------------------------------------
// Tiled bf16 WMMA GEMM:  out = epilogue(A[MxK] @ W[KxN] + bias)
//   W passed PRE-TRANSPOSED as Wt[N,K]; all staging/fragment traffic is
//   16B-aligned b128. BM=64 BN=256 BK=32, 256 threads = 8 waves (2x4),
//   wave tile 32x64 -> 8x v_wmma_f32_16x16x32_bf16 per wave per K-step.
// ---------------------------------------------------------------------------
#define BM 64
#define BN 256
#define BK 32
#define LDS_STRIDE 48   // elements; 96B rows keep uint4 LDS accesses 16B-aligned

template <bool OUT_BF16, bool DO_GELU, bool DO_RESID, bool DO_GATE>
__global__ __launch_bounds__(256) void gemm_bf16_wmma(
    const unsigned short* __restrict__ A, const unsigned short* __restrict__ Wt,
    const float* __restrict__ bias, const float* __restrict__ resid,
    const float* __restrict__ gate, void* __restrict__ outp, int M, int N, int K) {
  __shared__ unsigned short As[BM * LDS_STRIDE];   //  6 KB
  __shared__ unsigned short Bs[BN * LDS_STRIDE];   // 24 KB, Bs[n][k] K-contiguous

  const int tid  = threadIdx.x;
  const int lane = tid & 31;
  const int wave = tid >> 5;      // 0..7
  const int wm   = wave >> 2;     // 0..1  -> 32-row strip
  const int wn   = wave & 3;      // 0..3  -> 64-col strip
  const int half = lane >> 4;     // WMMA half-wave selector
  const int l15  = lane & 15;

  const int m0 = blockIdx.y * BM;
  const int n0 = blockIdx.x * BN;

  v8f acc[2][4];
#pragma unroll
  for (int mi = 0; mi < 2; ++mi)
#pragma unroll
    for (int ni = 0; ni < 4; ++ni)
#pragma unroll
      for (int i = 0; i < 8; ++i) acc[mi][ni][i] = 0.0f;

  const int arow = tid >> 2;        // 0..63
  const int acol = (tid & 3) * 8;   // 0,8,16,24

  for (int k0 = 0; k0 < K; k0 += BK) {
    // ---- stage A tile (64 m x 32 k) into LDS ----
    uint4 av = *reinterpret_cast<const uint4*>(&A[(size_t)(m0 + arow) * K + k0 + acol]);
    *reinterpret_cast<uint4*>(&As[arow * LDS_STRIDE + acol]) = av;
    // ---- stage B tile (256 n x 32 k) into LDS, K-contiguous (one row/thread) ----
    const unsigned short* wsrc = &Wt[(size_t)(n0 + tid) * K + k0];
    unsigned short* bdst = &Bs[tid * LDS_STRIDE];
#pragma unroll
    for (int c = 0; c < 4; ++c) {
      uint4 bv = *reinterpret_cast<const uint4*>(wsrc + c * 8);
      *reinterpret_cast<uint4*>(bdst + c * 8) = bv;
    }
    if (k0 + BK < K) {   // pull next K-tile toward near caches
      __builtin_prefetch(&A[(size_t)(m0 + arow) * K + k0 + BK + acol], 0, 3);
      __builtin_prefetch(wsrc + BK, 0, 3);
    }
    __syncthreads();

    // ---- build fragments per documented CDNA5 VGPR layouts ----
    union Frag { uint4 q[2]; v16bf v; };
    Frag afr[2], bfr[4];
#pragma unroll
    for (int mi = 0; mi < 2; ++mi) {
      const unsigned short* ap = &As[(wm * 32 + mi * 16 + l15) * LDS_STRIDE];
      afr[mi].q[0] = *reinterpret_cast<const uint4*>(ap + half * 8);        // K = kb..kb+7
      afr[mi].q[1] = *reinterpret_cast<const uint4*>(ap + 16 + half * 8);   // K = 16+kb..
    }
#pragma unroll
    for (int ni = 0; ni < 4; ++ni) {
      const unsigned short* bp = &Bs[(wn * 64 + ni * 16 + l15) * LDS_STRIDE];
      bfr[ni].q[0] = *reinterpret_cast<const uint4*>(bp + half * 8);
      bfr[ni].q[1] = *reinterpret_cast<const uint4*>(bp + 16 + half * 8);
    }
#pragma unroll
    for (int mi = 0; mi < 2; ++mi)
#pragma unroll
      for (int ni = 0; ni < 4; ++ni)
        acc[mi][ni] = __builtin_amdgcn_wmma_f32_16x16x32_bf16(
            false, afr[mi].v, false, bfr[ni].v, (short)0, acc[mi][ni], false, false);
    __syncthreads();
  }

  float gmul = 1.0f;
  if (DO_GATE) gmul = 1.0f / (1.0f + expf(-gate[0]));

#pragma unroll
  for (int mi = 0; mi < 2; ++mi) {
#pragma unroll
    for (int ni = 0; ni < 4; ++ni) {
      const int c = n0 + wn * 64 + ni * 16 + l15;
#pragma unroll
      for (int i = 0; i < 8; ++i) {
        const int r = m0 + wm * 32 + mi * 16 + half * 8 + i;
        float v = acc[mi][ni][i] + bias[c];
        if (DO_GELU) v = gelu_t(v);
        if (DO_RESID) v = resid[(size_t)r * N + c] + gmul * v;
        if (OUT_BF16)
          ((unsigned short*)outp)[(size_t)r * N + c] = f2bf_bits(v);
        else
          ((float*)outp)[(size_t)r * N + c] = v;
      }
    }
  }
}

// ---------------------------------------------------------------------------
// LayerNorm over 1024 columns; 256 threads/row; bf16 output
// ---------------------------------------------------------------------------
template <typename TIN>
__global__ __launch_bounds__(256) void layernorm_kernel(const TIN* __restrict__ x,
                                                        const float* __restrict__ g,
                                                        const float* __restrict__ b,
                                                        unsigned short* __restrict__ y) {
  const int row = blockIdx.x;
  const int tid = threadIdx.x;
  float v[4], s = 0.f, s2 = 0.f;
#pragma unroll
  for (int j = 0; j < 4; ++j) {
    v[j] = loadval(&x[(size_t)row * D_MODEL + tid + j * 256]);
    s += v[j]; s2 += v[j] * v[j];
  }
#pragma unroll
  for (int off = 16; off > 0; off >>= 1) {
    s  += __shfl_xor(s, off, 32);
    s2 += __shfl_xor(s2, off, 32);
  }
  __shared__ float red[16];
  __shared__ float stat[2];
  const int wid = tid >> 5;
  if ((tid & 31) == 0) { red[wid] = s; red[8 + wid] = s2; }
  __syncthreads();
  if (tid == 0) {
    float ts = 0.f, t2 = 0.f;
    for (int i = 0; i < 8; ++i) { ts += red[i]; t2 += red[8 + i]; }
    float mu = ts * (1.0f / D_MODEL);
    float var = t2 * (1.0f / D_MODEL) - mu * mu;
    stat[0] = mu; stat[1] = rsqrtf(var + 1e-5f);
  }
  __syncthreads();
  const float mu = stat[0], rs = stat[1];
#pragma unroll
  for (int j = 0; j < 4; ++j) {
    const int c = tid + j * 256;
    y[(size_t)row * D_MODEL + c] = f2bf_bits((v[j] - mu) * rs * g[c] + b[c]);
  }
}

// ---------------------------------------------------------------------------
// Cross attention: 1 query vs NT=4 kv tokens; 1 wave per head (wave32)
// ---------------------------------------------------------------------------
__global__ __launch_bounds__(128) void attn_kernel(const unsigned short* __restrict__ qh,
                                                   const unsigned short* __restrict__ kh,
                                                   const unsigned short* __restrict__ vh,
                                                   unsigned short* __restrict__ ctx) {
  const int b    = blockIdx.x;
  const int head = threadIdx.x >> 5;
  const int lane = threadIdx.x & 31;
  const size_t qbase = (size_t)b * D_MODEL + head * DH_HEAD + lane * 8;

  float qv[8];
  {
    uint4 q4 = *reinterpret_cast<const uint4*>(&qh[qbase]);
    const unsigned short* qs = reinterpret_cast<const unsigned short*>(&q4);
#pragma unroll
    for (int j = 0; j < 8; ++j) qv[j] = bf2f_bits(qs[j]);
  }

  float sc[NT_TOK];
#pragma unroll
  for (int t = 0; t < NT_TOK; ++t) {
    const size_t kbase = ((size_t)b * NT_TOK + t) * D_MODEL + head * DH_HEAD + lane * 8;
    uint4 k4 = *reinterpret_cast<const uint4*>(&kh[kbase]);
    const unsigned short* ks = reinterpret_cast<const unsigned short*>(&k4);
    float p = 0.f;
#pragma unroll
    for (int j = 0; j < 8; ++j) p += qv[j] * bf2f_bits(ks[j]);
#pragma unroll
    for (int off = 16; off > 0; off >>= 1) p += __shfl_xor(p, off, 32);
    sc[t] = p * (1.0f / 16.0f);   // 1/sqrt(DH=256)
  }

  float mx = fmaxf(fmaxf(sc[0], sc[1]), fmaxf(sc[2], sc[3]));
  float w[NT_TOK], sum = 0.f;
#pragma unroll
  for (int t = 0; t < NT_TOK; ++t) { w[t] = expf(sc[t] - mx); sum += w[t]; }
  const float inv = 1.0f / sum;
#pragma unroll
  for (int t = 0; t < NT_TOK; ++t) w[t] *= inv;

  float o[8] = {0.f, 0.f, 0.f, 0.f, 0.f, 0.f, 0.f, 0.f};
#pragma unroll
  for (int t = 0; t < NT_TOK; ++t) {
    const size_t vbase = ((size_t)b * NT_TOK + t) * D_MODEL + head * DH_HEAD + lane * 8;
    uint4 v4 = *reinterpret_cast<const uint4*>(&vh[vbase]);
    const unsigned short* vs = reinterpret_cast<const unsigned short*>(&v4);
#pragma unroll
    for (int j = 0; j < 8; ++j) o[j] += w[t] * bf2f_bits(vs[j]);
  }
#pragma unroll
  for (int j = 0; j < 8; ++j) ctx[qbase + j] = f2bf_bits(o[j]);
}

// ---------------------------------------------------------------------------
// Host-side orchestration
// ---------------------------------------------------------------------------
extern "C" void kernel_launch(void* const* d_in, const int* in_sizes, int n_in,
                              void* d_out, int out_size, void* d_ws, size_t ws_size,
                              hipStream_t stream) {
  (void)in_sizes; (void)n_in; (void)out_size; (void)ws_size;
  const float* embedding = (const float*)d_in[0];
  const float* physics   = (const float*)d_in[1];
  const float* pw1  = (const float*)d_in[2];
  const float* pb1  = (const float*)d_in[3];
  const float* pw2  = (const float*)d_in[4];
  const float* pb2  = (const float*)d_in[5];
  const float* lnq_g  = (const float*)d_in[6];
  const float* lnq_b  = (const float*)d_in[7];
  const float* lnkv_g = (const float*)d_in[8];
  const float* lnkv_b = (const float*)d_in[9];
  const float* wq = (const float*)d_in[10]; const float* bq = (const float*)d_in[11];
  const float* wk = (const float*)d_in[12]; const float* bk = (const float*)d_in[13];
  const float* wv = (const float*)d_in[14]; const float* bv = (const float*)d_in[15];
  const float* wo = (const float*)d_in[16]; const float* bo = (const float*)d_in[17];
  const float* ffn_g = (const float*)d_in[18]; const float* ffn_b = (const float*)d_in[19];
  const float* fw1 = (const float*)d_in[20]; const float* fb1 = (const float*)d_in[21];
  const float* fw2 = (const float*)d_in[22]; const float* fb2 = (const float*)d_in[23];
  const float* gate = (const float*)d_in[24];
  float* out = (float*)d_out;

  // ---- workspace layout (256B aligned slabs) ----
  char* base = (char*)d_ws;
  size_t off = 0;
  auto alloc = [&](size_t bytes) -> void* {
    off = (off + 255) & ~(size_t)255;
    void* p = base + off;
    off += bytes;
    return p;
  };
  unsigned short* h_bf  = (unsigned short*)alloc((size_t)B_TOK * DN_PROJ * 2); // reused as kh
  unsigned short* kvraw = (unsigned short*)alloc((size_t)B_TOK * DN_PROJ * 2); // reused as vh
  unsigned short* pw2b  = (unsigned short*)alloc((size_t)DN_PROJ * DN_PROJ * 2);
  unsigned short* wqb   = (unsigned short*)alloc((size_t)D_MODEL * D_MODEL * 2);
  unsigned short* wkb   = (unsigned short*)alloc((size_t)D_MODEL * D_MODEL * 2);
  unsigned short* wvb   = (unsigned short*)alloc((size_t)D_MODEL * D_MODEL * 2);
  unsigned short* wob   = (unsigned short*)alloc((size_t)D_MODEL * D_MODEL * 2);
  unsigned short* fw1b  = (unsigned short*)alloc((size_t)D_MODEL * 2 * D_MODEL * 2);
  unsigned short* fw2b  = (unsigned short*)alloc((size_t)2 * D_MODEL * D_MODEL * 2);
  unsigned short* qln   = (unsigned short*)alloc((size_t)B_TOK * D_MODEL * 2);
  unsigned short* kvln  = (unsigned short*)alloc((size_t)B_TOK * DN_PROJ * 2);
  unsigned short* qh    = (unsigned short*)alloc((size_t)B_TOK * D_MODEL * 2);
  unsigned short* ctx   = (unsigned short*)alloc((size_t)B_TOK * D_MODEL * 2);
  float*          fused = (float*)alloc((size_t)B_TOK * D_MODEL * 4);
  unsigned short* tln   = (unsigned short*)alloc((size_t)B_TOK * D_MODEL * 2);
  unsigned short* ub    = (unsigned short*)alloc((size_t)B_TOK * 2 * D_MODEL * 2);
  unsigned short* kh = h_bf;    // alias: h dead after big GEMM
  unsigned short* vh = kvraw;   // alias: raw kv dead after its LN

  const dim3 blk256(256), blk128(128);

  // 1) weight conversions fp32 [K,N] -> bf16 bits [N,K] (transposed once)
  f2bf_transpose_kernel<<<dim3(DN_PROJ / 32, DN_PROJ / 32), blk256, 0, stream>>>(pw2, pw2b, DN_PROJ, DN_PROJ);
  f2bf_transpose_kernel<<<dim3(D_MODEL / 32, D_MODEL / 32), blk256, 0, stream>>>(wq, wqb, D_MODEL, D_MODEL);
  f2bf_transpose_kernel<<<dim3(D_MODEL / 32, D_MODEL / 32), blk256, 0, stream>>>(wk, wkb, D_MODEL, D_MODEL);
  f2bf_transpose_kernel<<<dim3(D_MODEL / 32, D_MODEL / 32), blk256, 0, stream>>>(wv, wvb, D_MODEL, D_MODEL);
  f2bf_transpose_kernel<<<dim3(D_MODEL / 32, D_MODEL / 32), blk256, 0, stream>>>(wo, wob, D_MODEL, D_MODEL);
  f2bf_transpose_kernel<<<dim3(2 * D_MODEL / 32, D_MODEL / 32), blk256, 0, stream>>>(fw1, fw1b, D_MODEL, 2 * D_MODEL);
  f2bf_transpose_kernel<<<dim3(D_MODEL / 32, 2 * D_MODEL / 32), blk256, 0, stream>>>(fw2, fw2b, 2 * D_MODEL, D_MODEL);

  // 2) h = gelu(physics @ pw1 + pb1)
  phys_h_kernel<<<B_TOK, blk256, 0, stream>>>(physics, pw1, pb1, h_bf);

  // 3) phys_kv = h @ pw2 + pb2   (M=16384, N=4096, K=4096)
  gemm_bf16_wmma<true, false, false, false>
      <<<dim3(DN_PROJ / BN, B_TOK / BM), blk256, 0, stream>>>(
          h_bf, pw2b, pb2, nullptr, nullptr, kvraw, B_TOK, DN_PROJ, DN_PROJ);

  // 4) layernorms
  layernorm_kernel<float><<<B_TOK, blk256, 0, stream>>>(embedding, lnq_g, lnq_b, qln);
  layernorm_kernel<unsigned short><<<B_TOK * NT_TOK, blk256, 0, stream>>>(kvraw, lnkv_g, lnkv_b, kvln);

  // 5) Q/K/V projections
  gemm_bf16_wmma<true, false, false, false>
      <<<dim3(D_MODEL / BN, B_TOK / BM), blk256, 0, stream>>>(
          qln, wqb, bq, nullptr, nullptr, qh, B_TOK, D_MODEL, D_MODEL);
  gemm_bf16_wmma<true, false, false, false>
      <<<dim3(D_MODEL / BN, (B_TOK * NT_TOK) / BM), blk256, 0, stream>>>(
          kvln, wkb, bk, nullptr, nullptr, kh, B_TOK * NT_TOK, D_MODEL, D_MODEL);
  gemm_bf16_wmma<true, false, false, false>
      <<<dim3(D_MODEL / BN, (B_TOK * NT_TOK) / BM), blk256, 0, stream>>>(
          kvln, wvb, bv, nullptr, nullptr, vh, B_TOK * NT_TOK, D_MODEL, D_MODEL);

  // 6) softmax cross-attention -> ctx
  attn_kernel<<<B_TOK, blk128, 0, stream>>>(qh, kh, vh, ctx);

  // 7) fused = embedding + sigmoid(gate) * (ctx @ wo + bo)
  gemm_bf16_wmma<false, false, true, true>
      <<<dim3(D_MODEL / BN, B_TOK / BM), blk256, 0, stream>>>(
          ctx, wob, bo, embedding, gate, fused, B_TOK, D_MODEL, D_MODEL);

  // 8) FFN: t = LN(fused); u = gelu(t @ fw1 + fb1); out = fused + u @ fw2 + fb2
  layernorm_kernel<float><<<B_TOK, blk256, 0, stream>>>(fused, ffn_g, ffn_b, tln);
  gemm_bf16_wmma<true, true, false, false>
      <<<dim3((2 * D_MODEL) / BN, B_TOK / BM), blk256, 0, stream>>>(
          tln, fw1b, fb1, nullptr, nullptr, ub, B_TOK, 2 * D_MODEL, D_MODEL);
  gemm_bf16_wmma<false, false, true, false>
      <<<dim3(D_MODEL / BN, B_TOK / BM), blk256, 0, stream>>>(
          ub, fw2b, fb2, fused, nullptr, out, B_TOK, D_MODEL, 2 * D_MODEL);
}
